// ResidualRBFLayer_22308060136149
// MI455X (gfx1250) — matrix operand order repaired and checked by reading, not compiled, not run on hardware
//
#include <hip/hip_runtime.h>
#include <hip/hip_bf16.h>

// ---------------------------------------------------------------------------
// MI455X (gfx1250) fused residual-RBF prox layer.
// conv7x7 -> GRBF -> convT7x7 fused per 16x16 tile; both convs are f16-in /
// f32-acc WMMA GEMMs (v_wmma_f32_16x16x32_f16); RBF is v_exp_f32 VALU work.
// All LDS operands zero-padded so every fragment load is unconditional
// (no exec-mask branches, no NaN hazards into WMMA); fragments move as
// aligned 16/32-byte vector LDS accesses; wave-uniform RBF centers go
// through the scalar (SGPR) path instead of LDS.
// ---------------------------------------------------------------------------

typedef _Float16 v16h __attribute__((ext_vector_type(16)));
typedef _Float16 v8h  __attribute__((ext_vector_type(8)));
typedef float    v8f  __attribute__((ext_vector_type(8)));

#define F_TOT   48
#define NTAP    49
#define KPAD    64          // taps padded to 2 WMMA K-steps of 32
#define M_RBF   51
#define HH      256
#define WW      256
#define TILE    16
#define AREG    22          // TILE + 6 (a/z halo of 3 each side)
#define NPIX    (AREG*AREG) // 484
#define NPIXP   496         // padded to 31 chunks of 16
#define NCHUNK  31
#define XREG    28          // TILE + 12 (x halo of 6 each side)
#define XPITCH  29
#define XTROWS  31          // 3 extra zero rows so taps 49..63 read 0 in-bounds
#define XT_ELEMS (XTROWS*XPITCH + 1)   // 900 floats

// LDS layout (dynamic), all major arrays 64B-aligned:
constexpr size_t OFF_S    = 0;                                   // [496][64] f16
constexpr size_t OFF_AH   = OFF_S    + (size_t)NPIXP * 64 * 2;   // [496][32] f16
constexpr size_t OFF_WH   = OFF_AH   + (size_t)NPIXP * 32 * 2;   // [48][64]  f16
constexpr size_t OFF_WHF  = OFF_WH   + (size_t)48 * KPAD * 2;    // [3][64][32] f16
constexpr size_t OFF_RBFW = OFF_WHF  + (size_t)3 * 64 * 32 * 2;  // [48][52]  f32
constexpr size_t OFF_XT   = OFF_RBFW + (size_t)48 * 52 * 4;      // [31][29]  f32
constexpr size_t SMEM_BYTES = OFF_XT + (size_t)XT_ELEMS * 4;     // 127,248 B

// Async global->LDS copy (ASYNCcnt path), bypassing builtin AS sema.
// Hardware maps generic addr[31:0] -> LDS byte address.
__device__ __forceinline__ void async_g2lds_b32(const float* gsrc, float* ldst) {
  unsigned int lds_off = (unsigned int)(unsigned long long)ldst;
  asm volatile("global_load_async_to_lds_b32 %0, %1, off"
               :: "v"(lds_off), "v"(gsrc)
               : "memory");
}

__device__ __forceinline__ void wait_asynccnt0() {
#if __has_builtin(__builtin_amdgcn_s_wait_asynccnt)
  __builtin_amdgcn_s_wait_asynccnt(0);
#else
  asm volatile("s_wait_asynccnt 0" ::: "memory");
#endif
}

// ---------------------------------------------------------------------------
// Kernel 1: normalize conv weights, zero per-image sum accumulators.
// ---------------------------------------------------------------------------
__global__ void __launch_bounds__(64)
prep_kernel(const float* __restrict__ w, const float* __restrict__ scale_f,
            float* __restrict__ wnorm, float* __restrict__ sums) {
  int f = threadIdx.x;
  if (f < F_TOT) {
    float m = 0.f;
    for (int i = 0; i < NTAP; ++i) m += w[f * NTAP + i];
    m *= (1.f / 49.f);
    float n = 0.f;
    for (int i = 0; i < NTAP; ++i) { float d = w[f * NTAP + i] - m; n += d * d; }
    float inv = scale_f[f] / (sqrtf(n) + 1e-12f);
    for (int i = 0; i < NTAP; ++i) wnorm[f * NTAP + i] = (w[f * NTAP + i] - m) * inv;
  } else {
    sums[f - F_TOT] = 0.f;   // 16 per-image ||r||^2 accumulators
  }
}

// ---------------------------------------------------------------------------
// Kernel 2: fused conv -> GRBF -> convT, residual + block-reduced ||r||^2.
// One 16x16 output tile per workgroup, 256 threads (8 waves of 32).
// __launch_bounds__(256, 4): cap at 256 VGPRs/wave.
// ---------------------------------------------------------------------------
__global__ void __launch_bounds__(256, 4)
fused_main(const float* __restrict__ input, const float* __restrict__ net_input,
           const float* __restrict__ rbf_w, const float* __restrict__ rbf_c,
           const float* __restrict__ wnorm, float* __restrict__ r_out,
           float* __restrict__ sums) {
  extern __shared__ char smem[];
  _Float16* S    = (_Float16*)(smem + OFF_S);     // [496][64]  pixel-major
  _Float16* a_h  = (_Float16*)(smem + OFF_AH);    // [496][32]  pixel-major
  _Float16* wh   = (_Float16*)(smem + OFF_WH);    // [48][64]
  _Float16* whf  = (_Float16*)(smem + OFF_WHF);   // [3][64][32]
  float*    rbfw = (float*)(smem + OFF_RBFW);     // [48][52]
  float*    xt   = (float*)(smem + OFF_XT);       // [31][29] (+1 pad)

  const int tid  = threadIdx.x;
  const int lane = tid & 31;
  const int wv   = tid >> 5;
  const int col  = lane & 15;
  const int half = lane >> 4;

  const int img = blockIdx.x >> 8;
  const int tb  = blockIdx.x & 255;
  const int ox  = (tb >> 4) << 4;
  const int oy  = (tb & 15) << 4;

  // ---- kick off async x-tile load first (ASYNCcnt, no VGPR staging) ----
  const float* xin = input + (size_t)img * (HH * WW);
  for (int idx = tid; idx < XREG * XREG; idx += 256) {
    int rx = idx / XREG, ry = idx % XREG;
    int gx = ox - 6 + rx, gy = oy - 6 + ry;
    gx = (gx < 0) ? (-gx - 1) : gx;  gx = (gx > 255) ? (511 - gx) : gx;
    gy = (gy < 0) ? (-gy - 1) : gy;  gy = (gy > 255) ? (511 - gy) : gy;
    async_g2lds_b32(xin + gx * WW + gy, xt + rx * XPITCH + ry);
  }

  // ---- zero-fill pads + preload tables (overlaps async load) ----
  // xt rows 28..30 (+pad): indices [812, 900)
  if (tid < XT_ELEMS - XREG * XPITCH) xt[XREG * XPITCH + tid] = 0.f;
  // a_h k=16..31 zero for every pixel (phase-2 B upper half)
  {
    v8h z8 = {(_Float16)0.f, (_Float16)0.f, (_Float16)0.f, (_Float16)0.f,
              (_Float16)0.f, (_Float16)0.f, (_Float16)0.f, (_Float16)0.f};
    for (int idx = tid; idx < NPIXP * 2; idx += 256) {
      int p = idx >> 1, h = idx & 1;
      *(v8h*)(a_h + p * 32 + 16 + h * 8) = z8;
    }
  }
  for (int idx = tid; idx < F_TOT * M_RBF; idx += 256) {
    int f = idx / M_RBF, m = idx % M_RBF;
    rbfw[f * 52 + m] = rbf_w[idx];
  }
  for (int idx = tid; idx < F_TOT * KPAD; idx += 256) {
    int f = idx >> 6, t = idx & 63;
    wh[f * KPAD + t] = (_Float16)((t < NTAP) ? wnorm[f * NTAP + t] : 0.f);
  }
  for (int idx = tid; idx < 3 * 64 * 32; idx += 256) {
    int g = idx >> 11, rem = idx & 2047, t = rem >> 5, kf = rem & 31;
    float v = (kf < 16 && t < NTAP) ? wnorm[(g * 16 + kf) * NTAP + (48 - t)] : 0.f;
    whf[idx] = (_Float16)v;
  }

  wait_asynccnt0();
  __syncthreads();

  float convt = 0.f;

  for (int g = 0; g < 3; ++g) {          // 3 groups of 16 filters
    // A fragments for forward GEMM: two contiguous 8-element runs per lane.
    v16h afr0, afr1;
    {
      const _Float16* wrow = wh + (g * 16 + col) * KPAD;
      v8h lo0 = *(const v8h*)(wrow + half * 8);
      v8h hi0 = *(const v8h*)(wrow + 16 + half * 8);
      v8h lo1 = *(const v8h*)(wrow + 32 + half * 8);
      v8h hi1 = *(const v8h*)(wrow + 48 + half * 8);
      afr0 = __builtin_shufflevector(lo0, hi0, 0,1,2,3,4,5,6,7,8,9,10,11,12,13,14,15);
      afr1 = __builtin_shufflevector(lo1, hi1, 0,1,2,3,4,5,6,7,8,9,10,11,12,13,14,15);
    }

    // ---- phase 1: z = W * patches (WMMA), a = GRBF(z) -> LDS (f16) ----
    for (int ch = wv; ch < NCHUNK; ch += 8) {   // wave-uniform: EXEC stays full
      int pid  = ch * 16 + col;
      int p    = (pid < NPIX) ? pid : (NPIX - 1);
      int base = (p / AREG) * XPITCH + (p % AREG);
      v8f acc = {0.f, 0.f, 0.f, 0.f, 0.f, 0.f, 0.f, 0.f};
      #pragma unroll
      for (int kk = 0; kk < 2; ++kk) {
        // Unconditional loads: taps 49..63 hit the zeroed xt rows.
        v16h b;
        #pragma unroll
        for (int v = 0; v < 8; ++v) {
          int t0 = kk * 32 + half * 8 + 2 * v + (v & 4) * 2;
          float e0 = xt[base + (t0 / 7) * XPITCH + (t0 % 7)];
          float e1 = xt[base + ((t0 + 1) / 7) * XPITCH + ((t0 + 1) % 7)];
          b[2 * v]     = (_Float16)e0;
          b[2 * v + 1] = (_Float16)e1;
        }
        acc = __builtin_amdgcn_wmma_f32_16x16x32_f16(
            false, kk ? afr1 : afr0, false, b, (short)0, acc, false, false);
      }
      // GRBF on the 8 accumulator rows; unroll 4 on centers keeps VGPRs sane.
      // Centers are wave-uniform -> read from global so they ride the scalar
      // (s_load / SGPR-operand) path instead of per-lane LDS loads.
      v8h av;
      #pragma unroll
      for (int r = 0; r < 8; ++r) {
        float z  = acc[r];
        float xc = fminf(fmaxf(z, -100.f), 100.f);
        const float* wr = rbfw + (g * 16 + r + half * 8) * 52;
        float s = 0.f;
        #pragma unroll 4
        for (int m = 0; m < M_RBF; ++m) {
          float d = xc - rbf_c[m];
          s = fmaf(wr[m], __expf(-0.01f * d * d), s);   // v_exp_f32
        }
        av[r] = (_Float16)s;
      }
      *(v8h*)(a_h + pid * 32 + half * 8) = av;          // one 16B LDS store
    }
    __syncthreads();

    // ---- phase 2: S[t, p] = sum_f wflip[f,t] * a[f,p]  (WMMA, K=16 real) ----
    for (int job = wv; job < 4 * NCHUNK; job += 8) {
      int mt  = job / NCHUNK;
      int ch  = job % NCHUNK;
      int pid = ch * 16 + col;
      const _Float16* wr = whf + g * 2048 + (mt * 16 + col) * 32;
      v8h lo = *(const v8h*)(wr + half * 16);
      v8h hi = *(const v8h*)(wr + half * 16 + 8);
      v16h a2 = __builtin_shufflevector(lo, hi, 0,1,2,3,4,5,6,7,8,9,10,11,12,13,14,15);
      // B fragment: 16 contiguous f16 (k = half*16 .. +15); k>=16 rows are zero.
      v16h b2 = *(const v16h*)(a_h + pid * 32 + half * 16);
      v8f sacc = {0.f, 0.f, 0.f, 0.f, 0.f, 0.f, 0.f, 0.f};
      sacc = __builtin_amdgcn_wmma_f32_16x16x32_f16(
          false, a2, false, b2, (short)0, sacc, false, false);
      v8h sv;
      #pragma unroll
      for (int r = 0; r < 8; ++r) sv[r] = (_Float16)sacc[r];
      *(v8h*)(S + pid * 64 + mt * 16 + half * 8) = sv;  // one 16B LDS store
    }
    __syncthreads();

    // ---- phase 3: convT shift-gather over 49 taps ----
    {
      int lx = tid >> 4, ly = tid & 15;
      float accp = 0.f;
      #pragma unroll 7
      for (int t = 0; t < NTAP; ++t) {
        int i = t / 7, j = t % 7;
        accp += (float)S[((lx + i) * AREG + (ly + j)) * 64 + t];
      }
      convt += accp;
    }
    __syncthreads();
  }

  // ---- epilogue: residual + per-image ||r||^2 partial sum ----
  int lx = tid >> 4, ly = tid & 15;
  size_t gpix = (size_t)img * (HH * WW) + (ox + lx) * WW + (oy + ly);
  float rv = input[gpix] - convt - net_input[gpix];
  r_out[gpix] = rv;

  float* red = (float*)smem;         // reuse S region as f32 reduction scratch
  red[tid] = rv * rv;
  __syncthreads();
  #pragma unroll
  for (int s2 = 128; s2 > 0; s2 >>= 1) {
    if (tid < s2) red[tid] += red[tid + s2];
    __syncthreads();
  }
  if (tid == 0) atomicAdd(&sums[img], red[0]);   // global_atomic_add_f32
}

// ---------------------------------------------------------------------------
// Kernel 3: ball projection scale + final output.
// ---------------------------------------------------------------------------
__global__ void __launch_bounds__(256)
finish_kernel(const float* __restrict__ net_input, const float* __restrict__ r,
              const float* __restrict__ sums, const float* __restrict__ stdn,
              const float* __restrict__ alpha, float* __restrict__ out, int n) {
  int i = blockIdx.x * 256 + threadIdx.x;
  if (i >= n) return;
  int img = i >> 16;                                  // 65536 px per image
  float k  = __expf(alpha[0]) * stdn[img] * 256.f;    // sqrt(1*256*256) = 256
  float nr = sqrtf(sums[img]);
  float sc = fminf(1.f, k / (nr + 1e-12f));
  out[i] = net_input[i] + r[i] * sc;
}

// ---------------------------------------------------------------------------
extern "C" void kernel_launch(void* const* d_in, const int* in_sizes, int n_in,
                              void* d_out, int out_size, void* d_ws, size_t ws_size,
                              hipStream_t stream) {
  (void)in_sizes; (void)n_in; (void)out_size; (void)ws_size;

  const float* input   = (const float*)d_in[0];
  const float* stdn    = (const float*)d_in[1];
  // d_in[2] = rbf_data (unused by forward math)
  const float* net_in  = (const float*)d_in[3];
  const float* convw   = (const float*)d_in[4];
  const float* scale_f = (const float*)d_in[5];
  const float* alpha   = (const float*)d_in[6];
  const float* rbf_w   = (const float*)d_in[7];
  const float* rbf_c   = (const float*)d_in[8];

  // workspace: residual image (4 MB) + normalized weights + per-image sums
  float* ws_r    = (float*)d_ws;
  float* ws_wn   = ws_r + (size_t)16 * 65536;
  float* ws_sums = ws_wn + F_TOT * NTAP;

  prep_kernel<<<1, 64, 0, stream>>>(convw, scale_f, ws_wn, ws_sums);

  (void)hipFuncSetAttribute((const void*)fused_main,
                            hipFuncAttributeMaxDynamicSharedMemorySize,
                            (int)SMEM_BYTES);
  fused_main<<<dim3(16 * 256), dim3(256), SMEM_BYTES, stream>>>(
      input, net_in, rbf_w, rbf_c, ws_wn, ws_r, ws_sums);

  int n = 16 * 65536;
  finish_kernel<<<dim3((n + 255) / 256), dim3(256), 0, stream>>>(
      net_in, ws_r, ws_sums, stdn, alpha, (float*)d_out, n);
}
